// AttentionLayer_60387240182493
// MI455X (gfx1250) — matrix operand order repaired and checked by reading, not compiled, not run on hardware
//
#include <hip/hip_runtime.h>
#include <hip/hip_bf16.h>

// ---------------------------------------------------------------------------
// Fused attention for N=16384, C=64 on gfx1250 (wave32, WMMA f16 16x16x32).
//   q = query@Wq ; k = hidden@Wk ; v = hidden@Wv   (f32 -> f16, WMMA)
//   out = softmax(q k^T / 8) @ v                   (flash-attention, fused)
//
// S^T = K Q^T formulation: each lane owns one query column of the score tile,
// so softmax reductions are in-lane (+1 shfl_xor(16)) and the S^T accumulator
// layout coincides element-for-element with the f16 A-fragment layout of P
// (P@V feeds straight from registers). Double-buffered LDS staging with one
// barrier per key tile; rescale-skip when no new running max.
//
// K/V tiles staged with the Tensor Data Mover (tensor_load_to_lds, TENSORcnt)
// using the 6-arg builtin signature; fallback path kept for safety.
// ---------------------------------------------------------------------------

typedef __attribute__((ext_vector_type(16))) _Float16 v16h;
typedef __attribute__((ext_vector_type(8)))  float    v8f;
typedef unsigned int u32x4 __attribute__((ext_vector_type(4)));
typedef int          i32x8 __attribute__((ext_vector_type(8)));
typedef int          i32x4 __attribute__((ext_vector_type(4)));

#define NKEYS 16384
#define CDIM  64
#define NTILE (NKEYS / 32)

#define WMMA_F16(A, B, C) \
  __builtin_amdgcn_wmma_f32_16x16x32_f16(false, (A), false, (B), (short)0, (C), false, false)

// 0.125 * log2(e): softmax computed in exp2 domain (exact base change).
#define SCALE_LOG2E 0.1803368801111204f

#if __has_builtin(__builtin_amdgcn_tensor_load_to_lds) && \
    __has_builtin(__builtin_amdgcn_s_wait_tensorcnt)
#define USE_TDM 1
#else
#define USE_TDM 0
#endif

__device__ __forceinline__ v16h ld_pair(const _Float16* p0, const _Float16* p1) {
  v16h r;
  ((uint4*)&r)[0] = *(const uint4*)p0;
  ((uint4*)&r)[1] = *(const uint4*)p1;
  return r;
}

#if USE_TDM
// Issue a 2-D TDM tile load: tile (t0 x t1) elements (2-byte) from a
// row-major tensor (row length td0, row stride s0) into LDS at lds_addr.
// D# bitfields per CDNA5 ISA ch.8 (group0 128b, group1 256b; groups 2/3 zero).
__device__ __forceinline__ void tdm_load_2d(unsigned long long gaddr,
                                            unsigned int lds_addr,
                                            unsigned int td0, unsigned int td1,
                                            unsigned long long s0,
                                            unsigned int t0, unsigned int t1) {
  u32x4 g0;
  g0.x = 1u;                                        // count=1, user mode
  g0.y = lds_addr;                                  // lds_addr[31:0]
  g0.z = (unsigned int)gaddr;                       // global_addr[31:0]
  g0.w = (unsigned int)((gaddr >> 32) & 0x1FFFFFFu) // global_addr[56:32]
         | (2u << 30);                              // type = 2 ("image")
  i32x8 g1;
  g1[0] = (int)(1u << 16);                          // data_size=1 (2 bytes)
  g1[1] = (int)((td0 & 0xFFFFu) << 16);             // tensor_dim0[15:0] @bit48
  g1[2] = (int)((td0 >> 16) | ((td1 & 0xFFFFu) << 16));  // td0 hi | td1 lo
  g1[3] = (int)((td1 >> 16) | (t0 << 16));          // td1 hi | tile_dim0
  g1[4] = (int)(t1 & 0xFFFFu);                      // tile_dim1 (tile_dim2=0)
  g1[5] = (int)(unsigned int)(s0 & 0xFFFFFFFFu);    // dim0_stride[31:0]
  g1[6] = (int)(unsigned int)((s0 >> 32) & 0xFFFFu);// dim0_stride[47:32]
  g1[7] = 0;
  i32x4 z4 = {0, 0, 0, 0};
  i32x8 z8 = {0, 0, 0, 0, 0, 0, 0, 0};
  __builtin_amdgcn_tensor_load_to_lds(g0, g1, z4, z4, z8, 0);
}
#endif

// ---------------------------------------------------------------------------
// Projection: Y = X @ W  (X: [N,64] f32, W: [64,64] f32, Y: f16).
// TRANSPOSED=false -> Y row-major [N][64]; true -> Y = (X@W)^T as [64][N].
// ---------------------------------------------------------------------------
template <bool TRANSPOSED>
__global__ __launch_bounds__(256) void proj_kernel(const float* __restrict__ X,
                                                   const float* __restrict__ W,
                                                   _Float16* __restrict__ Y) {
  __shared__ __align__(16) _Float16 Wt[64][64];  // Wt[n][k] = W[k][n]
  const int t = threadIdx.x;
  for (int i = t; i < 64 * 64; i += 256) {
    int r = i >> 6, c = i & 63;
    Wt[c][r] = (_Float16)W[i];
  }
  __syncthreads();

  const int w = t >> 5, lane = t & 31;
  const int h = lane >> 4, mn = lane & 15;
  const int rb = blockIdx.x * 128 + w * 16;

  v16h af[2];  // A fragments of X rows (f32 -> f16), K chunks 0 and 32
#pragma unroll
  for (int c = 0; c < 2; ++c) {
    const float* p0 = X + (size_t)(rb + mn) * CDIM + 32 * c + 8 * h;
    const float* p1 = p0 + 16;
#pragma unroll
    for (int i = 0; i < 8; ++i) {
      af[c][i]     = (_Float16)p0[i];
      af[c][8 + i] = (_Float16)p1[i];
    }
  }

#pragma unroll
  for (int nt = 0; nt < 4; ++nt) {
    v8f acc = {0.f, 0.f, 0.f, 0.f, 0.f, 0.f, 0.f, 0.f};
#pragma unroll
    for (int c = 0; c < 2; ++c) {
      const _Float16* bp = &Wt[nt * 16 + mn][32 * c + 8 * h];
      v16h bf = ld_pair(bp, bp + 16);
      acc = WMMA_F16(af[c], bf, acc);
    }
    if (TRANSPOSED) {
      _Float16 tmp[8];
#pragma unroll
      for (int r = 0; r < 8; ++r) tmp[r] = (_Float16)acc[r];
      *(uint4*)&Y[(size_t)(nt * 16 + mn) * NKEYS + rb + 8 * h] = *(const uint4*)tmp;
    } else {
#pragma unroll
      for (int r = 0; r < 8; ++r)
        Y[(size_t)(rb + r + 8 * h) * CDIM + nt * 16 + mn] = (_Float16)acc[r];
    }
  }
}

// ---------------------------------------------------------------------------
// Flash attention. 8 waves / block, each wave owns a 16-query tile.
// Double-buffered LDS staging: one barrier per 32-key tile.
// ---------------------------------------------------------------------------
__global__ __launch_bounds__(256) void flash_kernel(const _Float16* __restrict__ Qh,
                                                    const _Float16* __restrict__ Kh,
                                                    const _Float16* __restrict__ Vt,
                                                    float* __restrict__ Out) {
  __shared__ __align__(16) _Float16 Kl[2][32 * 64];  // K tiles, row-major [key][c]
  __shared__ __align__(16) _Float16 Vl[2][64 * 32];  // V^T tiles [c][key]
  __shared__ __align__(16) float    abuf[8][16];     // per-wave alpha broadcast

  const int t = threadIdx.x;
  const int w = t >> 5, lane = t & 31;
  const int h = lane >> 4, mn = lane & 15;
  const int rb = blockIdx.x * 128 + w * 16;

  // Persistent Q as B-fragments: lane = query column, c in table order (8h, 16+8h).
  v16h qb[2];
#pragma unroll
  for (int c = 0; c < 2; ++c) {
    const _Float16* p = Qh + (size_t)(rb + mn) * CDIM + 32 * c + 8 * h;
    qb[c] = ld_pair(p, p + 16);
  }

  const v8f vz = {0.f, 0.f, 0.f, 0.f, 0.f, 0.f, 0.f, 0.f};
  v8f o[4];
  o[0] = vz; o[1] = vz; o[2] = vz; o[3] = vz;
  float mrow = -1e30f;  // running max (raw score domain), per-lane query
  float lrow = 0.f;     // running denominator, per-lane query

#if USE_TDM
  // TDM staging: wave 0 issues one DMA descriptor per matrix per tile.
  if (w == 0) {
    tdm_load_2d((unsigned long long)(uintptr_t)Kh,
                (unsigned int)(uintptr_t)&Kl[0][0], CDIM, NKEYS, CDIM, CDIM, 32);
    tdm_load_2d((unsigned long long)(uintptr_t)Vt,
                (unsigned int)(uintptr_t)&Vl[0][0], NKEYS, CDIM, NKEYS, 32, CDIM);
  }
#else
  // Fallback: per-thread register-pipelined staging (one 16B chunk each).
  const int kr = t >> 3, kc = (t & 7) * 8;
  const int vr = t >> 2, vc = (t & 3) * 8;
  const int klo = kr * 64 + kc;
  const int vlo = vr * 32 + vc;
  const _Float16* gk = Kh + (size_t)kr * CDIM + kc;
  const _Float16* gv = Vt + (size_t)vr * NKEYS + vc;
  uint4 kreg = *(const uint4*)gk;
  uint4 vreg = *(const uint4*)gv;
  *(uint4*)&Kl[0][klo] = kreg;
  *(uint4*)&Vl[0][vlo] = vreg;
  gk += 32 * CDIM;
  gv += 32;
  kreg = *(const uint4*)gk;
  vreg = *(const uint4*)gv;
#endif

  for (int i = 0; i < NTILE; ++i) {
    const int buf = i & 1;
#if USE_TDM
    if (w == 0) __builtin_amdgcn_s_wait_tensorcnt(0);  // tile i landed
#endif
    // Single barrier: (a) buf's data visible, (b) everyone done with buf^1.
    __syncthreads();
#if USE_TDM
    if (w == 0 && i + 1 < NTILE) {
      const size_t j1 = (size_t)(i + 1) * 32;
      tdm_load_2d((unsigned long long)(uintptr_t)(Kh + j1 * CDIM),
                  (unsigned int)(uintptr_t)&Kl[buf ^ 1][0],
                  CDIM, NKEYS - (unsigned)j1, CDIM, CDIM, 32);
      tdm_load_2d((unsigned long long)(uintptr_t)(Vt + j1),
                  (unsigned int)(uintptr_t)&Vl[buf ^ 1][0],
                  NKEYS - (unsigned)j1, CDIM, NKEYS, 32, CDIM);
    }
#else
    if (i + 1 < NTILE) {
      *(uint4*)&Kl[buf ^ 1][klo] = kreg;
      *(uint4*)&Vl[buf ^ 1][vlo] = vreg;
    }
    if (i + 2 < NTILE) {
      gk += 32 * CDIM;
      gv += 32;
      kreg = *(const uint4*)gk;  // prefetch tile i+2 during compute
      vreg = *(const uint4*)gv;
    }
#endif
    const _Float16* klb = &Kl[buf][0];
    const _Float16* vlb = &Vl[buf][0];

    // ---- S^T = K Q^T : two 16x16 accumulators (key tiles 0/1) ----
    v8f s0 = vz, s1 = vz;
#pragma unroll
    for (int cc = 0; cc < 2; ++cc) {
      const _Float16* a0 = &klb[mn * 64 + 32 * cc + 8 * h];
      v16h kf0 = ld_pair(a0, a0 + 16);
      s0 = WMMA_F16(kf0, qb[cc], s0);
      const _Float16* a1 = &klb[(16 + mn) * 64 + 32 * cc + 8 * h];
      v16h kf1 = ld_pair(a1, a1 + 16);
      s1 = WMMA_F16(kf1, qb[cc], s1);
    }

    // ---- online softmax, per-lane query (raw-domain max, exp2 eval) ----
    float tmv[8];
#pragma unroll
    for (int r = 0; r < 8; ++r) tmv[r] = fmaxf(s0[r], s1[r]);
#pragma unroll
    for (int st = 4; st >= 1; st >>= 1) {
#pragma unroll
      for (int r = 0; r < st; ++r) tmv[r] = fmaxf(tmv[r], tmv[r + st]);
    }
    float tmax = fmaxf(tmv[0], __shfl_xor(tmv[0], 16, 32));

    // Rescale only when some lane sees a new running max (rare after warmup).
    if (__any(tmax > mrow)) {
      float nm = fmaxf(mrow, tmax);
      float alpha = __builtin_amdgcn_exp2f((mrow - nm) * SCALE_LOG2E);
      mrow = nm;
      lrow *= alpha;
      if (h == 0) abuf[w][mn] = alpha;  // broadcast to C-layout rows
      float4 ar0 = *(const float4*)&abuf[w][8 * h];
      float4 ar1 = *(const float4*)&abuf[w][8 * h + 4];
      float arow[8] = {ar0.x, ar0.y, ar0.z, ar0.w, ar1.x, ar1.y, ar1.z, ar1.w};
#pragma unroll
      for (int q = 0; q < 4; ++q) {
#pragma unroll
        for (int r = 0; r < 8; ++r) o[q][r] = o[q][r] * arow[r];
      }
    }

    const float nb = -mrow * SCALE_LOG2E;
    float rs = 0.f;
#pragma unroll
    for (int r = 0; r < 8; ++r) {
      s0[r] = __builtin_amdgcn_exp2f(fmaf(s0[r], SCALE_LOG2E, nb));
      s1[r] = __builtin_amdgcn_exp2f(fmaf(s1[r], SCALE_LOG2E, nb));
      rs += s0[r] + s1[r];
    }
    rs += __shfl_xor(rs, 16, 32);
    lrow += rs;

    // ---- P A-fragment: pure in-register (S^T C-layout == P A-layout) ----
    v16h pa;
#pragma unroll
    for (int q = 0; q < 8; ++q) {
      pa[q]     = (_Float16)s0[q];
      pa[8 + q] = (_Float16)s1[q];
    }

    // ---- O += P @ V ----
#pragma unroll
    for (int nt = 0; nt < 4; ++nt) {
      const _Float16* vp = &vlb[(nt * 16 + mn) * 32 + 8 * h];
      v16h vb = ld_pair(vp, vp + 16);
      o[nt] = WMMA_F16(pa, vb, o[nt]);
    }
  }

  // ---- normalize (broadcast 1/l to C-layout rows) and store f32 ----
  float linv = 1.f / lrow;
  if (h == 0) abuf[w][mn] = linv;
  float4 iv0 = *(const float4*)&abuf[w][8 * h];
  float4 iv1 = *(const float4*)&abuf[w][8 * h + 4];
  float ivr[8] = {iv0.x, iv0.y, iv0.z, iv0.w, iv1.x, iv1.y, iv1.z, iv1.w};
#pragma unroll
  for (int nt = 0; nt < 4; ++nt) {
#pragma unroll
    for (int r = 0; r < 8; ++r)
      Out[(size_t)(rb + r + 8 * h) * CDIM + nt * 16 + mn] = o[nt][r] * ivr[r];
  }
}

// ---------------------------------------------------------------------------
extern "C" void kernel_launch(void* const* d_in, const int* in_sizes, int n_in,
                              void* d_out, int out_size, void* d_ws, size_t ws_size,
                              hipStream_t stream) {
  const float* hidden = (const float*)d_in[0];
  const float* query  = (const float*)d_in[1];
  const float* Wq     = (const float*)d_in[2];
  const float* Wk     = (const float*)d_in[3];
  const float* Wv     = (const float*)d_in[4];
  float* out = (float*)d_out;

  _Float16* Qh = (_Float16*)d_ws;                 // [N][64] f16
  _Float16* Kh = Qh + (size_t)NKEYS * CDIM;       // [N][64] f16
  _Float16* Vt = Kh + (size_t)NKEYS * CDIM;       // [64][N] f16

  dim3 grid(NKEYS / 128), blk(256);
  hipLaunchKernelGGL((proj_kernel<false>), grid, blk, 0, stream, query,  Wq, Qh);
  hipLaunchKernelGGL((proj_kernel<false>), grid, blk, 0, stream, hidden, Wk, Kh);
  hipLaunchKernelGGL((proj_kernel<true>),  grid, blk, 0, stream, hidden, Wv, Vt);
  hipLaunchKernelGGL(flash_kernel, grid, blk, 0, stream, Qh, Kh, Vt, out);
}
